// HyMRformer_80264348828122
// MI455X (gfx1250) — compile-verified, matrix operand order checked by reading, not run on hardware
//
#include <hip/hip_runtime.h>
#include <hip/hip_bf16.h>

// ---------- types ----------
typedef __attribute__((ext_vector_type(16))) __bf16 v16bf;
typedef __attribute__((ext_vector_type(8)))  float  v8f;

__device__ __forceinline__ unsigned short bfbits(float f) {
  unsigned u = __builtin_bit_cast(unsigned, f);
  u += 0x7FFFu + ((u >> 16) & 1u);            // round-to-nearest-even
  return (unsigned short)(u >> 16);
}

// ---------- constants ----------
#define N_ENT 200000
#define D_    400
#define B_    512
#define H_    4
#define EPS_  1e-5f

#define BM 128
#define BN 128
#define BK 32
#define LDP 40   // padded LDS row stride (bf16 elems); c4*2 stays 8B-aligned

// =====================================================================
// Generic GEMM: C[m,n] = sum_k A[m,k]*Bw[n,k] + bias[n]  (optional relu)
// A: M x K (row stride lda), Bw: N x K (row stride ldb)
// C stored at C[m*csm + n*csn].  bf16 WMMA with f32 accumulation.
// REQUIRES: K % 4 == 0, A/Bw rows 16B-aligned (lda/ldb % 4 == 0).
// Double-buffered LDS; fill is two-phase (batch 8x global_load_b128 into
// registers, then convert+store) so load latency overlaps the WMMAs.
// =====================================================================
__global__ void __launch_bounds__(256)
gemm_bf16_wmma_kernel(const float* __restrict__ A, const float* __restrict__ Bw,
                      const float* __restrict__ bias, float* __restrict__ C,
                      int M, int N, int K, int lda, int ldb,
                      int csm, int csn, int relu)
{
  __shared__ __bf16 As[2][BM][LDP];
  __shared__ __bf16 Bs[2][BN][LDP];

  const int tid   = threadIdx.x;
  const int lane  = tid & 31;
  const int wid   = tid >> 5;      // 0..7
  const int wr    = wid >> 1;      // 0..3 : wave row (32 rows each)
  const int wc    = wid & 1;       // 0..1 : wave col (64 cols each)
  const int lhalf = lane >> 4;     // 0/1
  const int lm    = lane & 15;

  const int m0 = blockIdx.y * BM;
  const int n0 = blockIdx.x * BN;

  v8f acc[2][4];
  v8f vz = {};
#pragma unroll
  for (int i = 0; i < 2; ++i)
#pragma unroll
    for (int j = 0; j < 4; ++j) acc[i][j] = vz;

  // ---- two-phase tile fill: batch clamped float4 loads, then pack+store
  constexpr int AIT = BM * (BK / 4) / 256;   // 4
  constexpr int BIT = BN * (BK / 4) / 256;   // 4
  auto fill = [&](int k0, int buf) {
    float4 va[AIT], vb4[BIT];
    bool   oka[AIT], okb[BIT];
    // phase 1: issue every global load (branch-free, clamped addresses)
#pragma unroll
    for (int u = 0; u < AIT; ++u) {
      int i  = tid + u * 256;
      int r  = i >> 3;
      int c4 = (i & 7) << 2;
      int gm = m0 + r, gk = k0 + c4;
      int gmc = gm < M ? gm : M - 1;
      int gkc = (gk + 3 < K) ? gk : K - 4;
      oka[u] = (gm < M) & (gk + 3 < K);
      va[u]  = *(const float4*)(A + (size_t)gmc * lda + gkc);
    }
#pragma unroll
    for (int u = 0; u < BIT; ++u) {
      int i  = tid + u * 256;
      int r  = i >> 3;
      int c4 = (i & 7) << 2;
      int gn = n0 + r, gk = k0 + c4;
      int gnc = gn < N ? gn : N - 1;
      int gkc = (gk + 3 < K) ? gk : K - 4;
      okb[u] = (gn < N) & (gk + 3 < K);
      vb4[u] = *(const float4*)(Bw + (size_t)gnc * ldb + gkc);
    }
    // phase 2: zero-select, pack 4xbf16, 8B LDS stores
#pragma unroll
    for (int u = 0; u < AIT; ++u) {
      int i  = tid + u * 256;
      int r  = i >> 3;
      int c4 = (i & 7) << 2;
      float4 v = oka[u] ? va[u] : make_float4(0.f, 0.f, 0.f, 0.f);
      uint2 p;
      p.x = (unsigned)bfbits(v.x) | ((unsigned)bfbits(v.y) << 16);
      p.y = (unsigned)bfbits(v.z) | ((unsigned)bfbits(v.w) << 16);
      *(uint2*)(&As[buf][r][c4]) = p;
    }
#pragma unroll
    for (int u = 0; u < BIT; ++u) {
      int i  = tid + u * 256;
      int r  = i >> 3;
      int c4 = (i & 7) << 2;
      float4 v = okb[u] ? vb4[u] : make_float4(0.f, 0.f, 0.f, 0.f);
      uint2 p;
      p.x = (unsigned)bfbits(v.x) | ((unsigned)bfbits(v.y) << 16);
      p.y = (unsigned)bfbits(v.z) | ((unsigned)bfbits(v.w) << 16);
      *(uint2*)(&Bs[buf][r][c4]) = p;
    }
  };

  const int nk = (K + BK - 1) / BK;
  fill(0, 0);

  for (int t = 0; t < nk; ++t) {
    __syncthreads();                 // buf(t) ready; buf(t^1) free to refill
    if (t + 1 < nk) fill((t + 1) * BK, (t + 1) & 1);
    if (t + 2 < nk) {                // prefetch tile after next into L2
      int gm = m0 + (tid >> 1);
      if (gm < M) __builtin_prefetch(&A[(size_t)gm * lda + (t + 2) * BK], 0, 0);
      int gn = n0 + (tid >> 1);
      if (gn < N) __builtin_prefetch(&Bw[(size_t)gn * ldb + (t + 2) * BK], 0, 0);
    }
    const int cb = t & 1;

    // fragments per CDNA5 16-bit A/B VGPR layout:
    // element e of lane -> k_local = (e/8)*16 + lhalf*8 + (e%8)
    v16bf afr[2], bfr[4];
#pragma unroll
    for (int mi = 0; mi < 2; ++mi) {
      const __bf16* ar = &As[cb][wr * 32 + mi * 16 + lm][0];
#pragma unroll
      for (int e = 0; e < 16; ++e)
        afr[mi][e] = ar[((e >> 3) << 4) + (lhalf << 3) + (e & 7)];
    }
#pragma unroll
    for (int ni = 0; ni < 4; ++ni) {
      const __bf16* br = &Bs[cb][wc * 64 + ni * 16 + lm][0];
#pragma unroll
      for (int e = 0; e < 16; ++e)
        bfr[ni][e] = br[((e >> 3) << 4) + (lhalf << 3) + (e & 7)];
    }

#pragma unroll
    for (int mi = 0; mi < 2; ++mi)
#pragma unroll
      for (int ni = 0; ni < 4; ++ni)
        acc[mi][ni] = __builtin_amdgcn_wmma_f32_16x16x32_bf16(
            false, afr[mi], false, bfr[ni], (short)0, acc[mi][ni], false, false);
  }

  // store: D f32 layout -> lane n = lm, vgpr r holds m = r + lhalf*8
#pragma unroll
  for (int mi = 0; mi < 2; ++mi) {
    int mb = m0 + wr * 32 + mi * 16 + lhalf * 8;
#pragma unroll
    for (int ni = 0; ni < 4; ++ni) {
      int n = n0 + wc * 64 + ni * 16 + lm;
      if (n >= N) continue;
      float bv = bias ? bias[n] : 0.f;
#pragma unroll
      for (int r = 0; r < 8; ++r) {
        int m = mb + r;
        if (m >= M) continue;
        float v = acc[mi][ni][r] + bv;
        if (relu) v = fmaxf(v, 0.f);
        C[(size_t)m * (size_t)csm + (size_t)n * (size_t)csn] = v;
      }
    }
  }
}

// =====================================================================
// x0[b,t,d] = (t==0 ? rel_emb[rel_idx[b]] : ent_emb[ent_idx[b,t-1]])[d]
//             + pos_emb[t,d]
// =====================================================================
__global__ void gather_x0_kernel(const float* __restrict__ ent,
                                 const float* __restrict__ rel,
                                 const float* __restrict__ pos,
                                 const int* __restrict__ rel_idx,
                                 const int* __restrict__ ent_idx,
                                 float* __restrict__ x0)
{
  int i = blockIdx.x * 256 + threadIdx.x;
  if (i >= B_ * 4 * D_) return;
  int d = i % D_;
  int t = (i / D_) & 3;
  int b = i / (4 * D_);
  float v;
  if (t == 0) v = rel[(size_t)rel_idx[b] * D_ + d];
  else        v = ent[(size_t)ent_idx[b * 3 + (t - 1)] * D_ + d];
  x0[i] = v + pos[t * D_ + d];
}

// =====================================================================
// tiny 4x4 attention per (b,h).  q/k/v: (B,4,1600) with [h*400+d]
// =====================================================================
__global__ void attn_kernel(const float* __restrict__ q,
                            const float* __restrict__ k,
                            const float* __restrict__ v,
                            float* __restrict__ o)
{
  int b = blockIdx.x, h = blockIdx.y;
  __shared__ float s[16];
  __shared__ float att[16];
  int tid = threadIdx.x;
  const float scale = 0.05f; // 1/sqrt(400)

  if (tid < 16) {
    int l = tid >> 2, m = tid & 3;
    const float* qp = q + ((size_t)(b * 4 + l)) * 1600 + h * D_;
    const float* kp = k + ((size_t)(b * 4 + m)) * 1600 + h * D_;
    float acc = 0.f;
    for (int d = 0; d < D_; ++d) acc += qp[d] * kp[d];
    s[tid] = acc * scale;
  }
  __syncthreads();
  if (tid < 4) {
    int l = tid;
    float mx = s[l * 4];
    for (int m = 1; m < 4; ++m) mx = fmaxf(mx, s[l * 4 + m]);
    float e[4], sum = 0.f;
    for (int m = 0; m < 4; ++m) { e[m] = __expf(s[l * 4 + m] - mx); sum += e[m]; }
    float inv = 1.f / sum;
    for (int m = 0; m < 4; ++m) att[l * 4 + m] = e[m] * inv;
  }
  __syncthreads();
  for (int idx = tid; idx < 4 * D_; idx += blockDim.x) {
    int l = idx / D_, d = idx % D_;
    float acc = 0.f;
    for (int m = 0; m < 4; ++m)
      acc += att[l * 4 + m] * v[((size_t)(b * 4 + m)) * 1600 + h * D_ + d];
    o[((size_t)(b * 4 + l)) * 1600 + h * D_ + d] = acc;
  }
}

// =====================================================================
// out[row] = LayerNorm(a[row] + dd[row]) * g + beta   (row of 400)
// =====================================================================
__global__ void add_ln_kernel(const float* __restrict__ a,
                              const float* __restrict__ dd,
                              const float* __restrict__ g,
                              const float* __restrict__ be,
                              float* __restrict__ out)
{
  int row = blockIdx.x;
  size_t base = (size_t)row * D_;
  __shared__ float rs[128], rq[128];
  float s = 0.f, q = 0.f;
  for (int c = threadIdx.x; c < D_; c += 128) {
    float v = a[base + c] + dd[base + c];
    s += v; q += v * v;
  }
  rs[threadIdx.x] = s; rq[threadIdx.x] = q;
  __syncthreads();
  for (int off = 64; off; off >>= 1) {
    if (threadIdx.x < off) {
      rs[threadIdx.x] += rs[threadIdx.x + off];
      rq[threadIdx.x] += rq[threadIdx.x + off];
    }
    __syncthreads();
  }
  float mean = rs[0] / (float)D_;
  float var  = rq[0] / (float)D_ - mean * mean;
  float rstd = rsqrtf(var + EPS_);
  for (int c = threadIdx.x; c < D_; c += 128) {
    float v = a[base + c] + dd[base + c];
    out[base + c] = (v - mean) * rstd * g[c] + be[c];
  }
}

// =====================================================================
// small dense: out[b,j] = sum_d A[b*lda + d]*W[j,d] + bias[j]
// =====================================================================
__global__ void small_lin_kernel(const float* __restrict__ A, int lda,
                                 const float* __restrict__ W,
                                 const float* __restrict__ bias,
                                 float* __restrict__ out, int N, int K)
{
  int b = blockIdx.x;
  int j = threadIdx.x;
  if (j >= N) return;
  const float* ar = A + (size_t)b * lda;
  const float* wr = W + (size_t)j * K;
  float acc = bias[j];
  for (int d = 0; d < K; ++d) acc += ar[d] * wr[d];
  out[(size_t)b * N + j] = acc;
}

// per-slot scalar mean/var over x2[:, s+1, :]  (grid = 3)
__global__ void stat_slots_kernel(const float* __restrict__ x2, float* __restrict__ st)
{
  int s = blockIdx.x;
  __shared__ float rs[256], rq[256];
  float sum = 0.f, sq = 0.f;
  for (int i = threadIdx.x; i < B_ * D_; i += 256) {
    int b = i / D_, d = i % D_;
    float v = x2[(size_t)(b * 4 + s + 1) * D_ + d];
    sum += v; sq += v * v;
  }
  rs[threadIdx.x] = sum; rq[threadIdx.x] = sq;
  __syncthreads();
  for (int off = 128; off; off >>= 1) {
    if (threadIdx.x < off) {
      rs[threadIdx.x] += rs[threadIdx.x + off];
      rq[threadIdx.x] += rq[threadIdx.x + off];
    }
    __syncthreads();
  }
  if (threadIdx.x == 0) {
    float n = (float)(B_ * D_);
    float m = rs[0] / n;
    st[2 * s] = m;
    st[2 * s + 1] = rq[0] / n - m * m;
  }
}

// scalar mean/var over a flat buffer (grid = 1)
__global__ void stat_flat_kernel(const float* __restrict__ p, int n, float* __restrict__ st2)
{
  __shared__ float rs[256], rq[256];
  float sum = 0.f, sq = 0.f;
  for (int i = threadIdx.x; i < n; i += 256) {
    float v = p[i]; sum += v; sq += v * v;
  }
  rs[threadIdx.x] = sum; rq[threadIdx.x] = sq;
  __syncthreads();
  for (int off = 128; off; off >>= 1) {
    if (threadIdx.x < off) {
      rs[threadIdx.x] += rs[threadIdx.x + off];
      rq[threadIdx.x] += rq[threadIdx.x + off];
    }
    __syncthreads();
  }
  if (threadIdx.x == 0) {
    float m = rs[0] / (float)n;
    st2[0] = m;
    st2[1] = rq[0] / (float)n - m * m;
  }
}

// y[(s*512+b)*6400 + p*400 + d] = f[b,p] * posbn(x2[b,s+1,d])
__global__ void build_y_kernel(const float* __restrict__ f,
                               const float* __restrict__ x2,
                               const float* __restrict__ st,
                               const float* __restrict__ pg,
                               const float* __restrict__ pb,
                               float* __restrict__ y)
{
  int i = blockIdx.x * 256 + threadIdx.x;
  if (i >= 3 * B_ * 6400) return;
  int d = i % D_;
  int p = (i / D_) & 15;
  int b = (i / 6400) & (B_ - 1);
  int s = i / (6400 * B_);
  float m = st[2 * s], v = st[2 * s + 1];
  float rstd = rsqrtf(v + EPS_);
  float xn = (x2[(size_t)(b * 4 + s + 1) * D_ + d] - m) * rstd * pg[0] + pb[0];
  y[i] = f[b * 16 + p] * xn;
}

// conv (3-tap over flattened ef) + pairwise maxpool
// ef stored as ef[s][b][d]; xr flat index idx=i*3+j -> slot idx/400, d idx%400
__global__ void conv_pool_kernel(const float* __restrict__ kern,
                                 const float* __restrict__ ef,
                                 const float* __restrict__ st,
                                 const float* __restrict__ bg,
                                 const float* __restrict__ bb,
                                 float* __restrict__ pooled)
{
  int b = blockIdx.x;
  __shared__ float ks[96];
  if (threadIdx.x < 96) ks[threadIdx.x] = kern[(size_t)b * 96 + threadIdx.x];
  __syncthreads();
  float m = st[0], v = st[1];
  float rstd = rsqrtf(v + EPS_);
  float g = bg[0], be = bb[0];
  for (int o = threadIdx.x; o < 32 * 200; o += 256) {
    int kk = o / 200, i2 = o % 200;
    float c[2];
#pragma unroll
    for (int t = 0; t < 2; ++t) {
      int i = 2 * i2 + t;
      float acc = 0.f;
#pragma unroll
      for (int j = 0; j < 3; ++j) {
        int idx = i * 3 + j;
        int s = idx / D_, d = idx - s * D_;
        float e = ef[((size_t)s * B_ + b) * D_ + d];
        acc += ks[kk * 3 + j] * ((e - m) * rstd * g + be);
      }
      c[t] = acc;
    }
    pooled[(size_t)b * 6400 + kk * 200 + i2] = fmaxf(c[0], c[1]);
  }
}

// =====================================================================
extern "C" void kernel_launch(void* const* d_in, const int* in_sizes, int n_in,
                              void* d_out, int out_size, void* d_ws, size_t ws_size,
                              hipStream_t stream)
{
  (void)in_sizes; (void)n_in; (void)out_size; (void)ws_size;

  const float* ent_emb     = (const float*)d_in[0];
  const float* rel_emb     = (const float*)d_in[1];
  const float* pos_emb     = (const float*)d_in[2];
  const float* relposw_w   = (const float*)d_in[3];
  const float* relposw_b   = (const float*)d_in[4];
  const float* relposinv_w = (const float*)d_in[5];
  const float* relposinv_b = (const float*)d_in[6];
  const float* relw4_w     = (const float*)d_in[7];
  const float* relw4_b     = (const float*)d_in[8];
  const float* q_w  = (const float*)d_in[9];
  const float* q_b  = (const float*)d_in[10];
  const float* k_w  = (const float*)d_in[11];
  const float* k_b  = (const float*)d_in[12];
  const float* v_w  = (const float*)d_in[13];
  const float* v_b  = (const float*)d_in[14];
  const float* fca_w = (const float*)d_in[15];
  const float* fca_b = (const float*)d_in[16];
  const float* ln1_g = (const float*)d_in[17];
  const float* ln1_b = (const float*)d_in[18];
  const float* ln2_g = (const float*)d_in[19];
  const float* ln2_b = (const float*)d_in[20];
  const float* w1_w  = (const float*)d_in[21];
  const float* w1_b  = (const float*)d_in[22];
  const float* w2_w  = (const float*)d_in[23];
  const float* w2_b  = (const float*)d_in[24];
  const float* fc_w  = (const float*)d_in[25];
  const float* fc_b  = (const float*)d_in[26];
  const float* posbn_g = (const float*)d_in[27];
  const float* posbn_b = (const float*)d_in[28];
  const float* bn_g  = (const float*)d_in[29];
  const float* bn_b  = (const float*)d_in[30];
  const float* biasv = (const float*)d_in[31];
  const int* rel_idx = (const int*)d_in[32];
  const int* ent_idx = (const int*)d_in[33];
  float* out = (float*)d_out;

  // ---- workspace layout (bytes) with aliasing ----
  char* w = (char*)d_ws;
  float* x0  = (float*)(w + 0);          //  3,276,800  (2048 x 400)
  float* qb  = (float*)(w + 3276800);    // 13,107,200  (2048 x 1600)
  float* kb  = (float*)(w + 16384000);   // 13,107,200
  float* vb  = (float*)(w + 29491200);   // 13,107,200
  float* ob  = (float*)(w + 42598400);   // 13,107,200 ; pooled aliases later
  float* yb  = qb;                       // 39,321,600 aliases q,k,v (dead after attn)
  float* t1  = (float*)(w + 55705600);   //  3,276,800
  float* x1  = (float*)(w + 58982400);   //  3,276,800
  float* hm  = (float*)(w + 62259200);   //  6,553,600  (2048 x 800)
  float* x2  = (float*)(w + 68812800);   //  3,276,800
  float* fb  = (float*)(w + 72089600);   //     32,768  (512 x 16)
  float* efb = (float*)(w + 72122368);   //  2,457,600  (3 x 512 x 400)
  float* knb = (float*)(w + 74579968);   //    196,608  (512 x 96)
  float* hb  = (float*)(w + 74776576);   //    819,200  (512 x 400)
  float* st  = (float*)(w + 75595776);   //  stats: [0..5] slot mean/var, [6..7] ef
  float* pooled = ob;

  auto gemm = [&](const float* A, const float* Bw, const float* bias, float* C,
                  int M, int N, int K, int lda, int ldb, int csm, int csn, int relu) {
    dim3 g((N + BN - 1) / BN, (M + BM - 1) / BM);
    gemm_bf16_wmma_kernel<<<g, dim3(256), 0, stream>>>(A, Bw, bias, C, M, N, K,
                                                       lda, ldb, csm, csn, relu);
  };

  // 1. gather + pos embed -> x0 (2048 x 400)
  gather_x0_kernel<<<(B_ * 4 * D_ + 255) / 256, 256, 0, stream>>>(
      ent_emb, rel_emb, pos_emb, rel_idx, ent_idx, x0);

  // 2. QKV projections (2048 x 1600, K=400)
  gemm(x0, q_w, q_b, qb, 2048, 1600, 400, 400, 400, 1600, 1, 0);
  gemm(x0, k_w, k_b, kb, 2048, 1600, 400, 400, 400, 1600, 1, 0);
  gemm(x0, v_w, v_b, vb, 2048, 1600, 400, 400, 400, 1600, 1, 0);

  // 3. attention (4x4 per (b,h))
  attn_kernel<<<dim3(B_, H_), 128, 0, stream>>>(qb, kb, vb, ob);

  // 4. fca + residual LN -> x1
  gemm(ob, fca_w, fca_b, t1, 2048, 400, 1600, 1600, 1600, 400, 1, 0);
  add_ln_kernel<<<2048, 128, 0, stream>>>(x0, t1, ln1_g, ln1_b, x1);

  // 5. FFN + residual LN -> x2
  gemm(x1, w1_w, w1_b, hm, 2048, 800, 400, 400, 400, 800, 1, 1);
  gemm(hm, w2_w, w2_b, t1, 2048, 400, 800, 800, 800, 400, 1, 0);
  add_ln_kernel<<<2048, 128, 0, stream>>>(x1, t1, ln2_g, ln2_b, x2);

  // 6. f = r @ relposw^T + b   (r = x2[:,0,:], row stride 1600)
  small_lin_kernel<<<B_, 32, 0, stream>>>(x2, 1600, relposw_w, relposw_b, fb, 16, 400);

  // 7. per-slot batch stats, build y, relposinv GEMM -> ef (3 slots fused M=1536)
  stat_slots_kernel<<<3, 256, 0, stream>>>(x2, st);
  build_y_kernel<<<(3 * B_ * 6400 + 255) / 256, 256, 0, stream>>>(
      fb, x2, st, posbn_g, posbn_b, yb);
  gemm(yb, relposinv_w, relposinv_b, efb, 1536, 400, 6400, 6400, 6400, 400, 1, 0);

  // 8. kern = r @ relw4^T + b  (512 x 96)
  small_lin_kernel<<<B_, 128, 0, stream>>>(x2, 1600, relw4_w, relw4_b, knb, 96, 400);

  // 9. ef global stats, conv + maxpool -> pooled (512 x 6400)
  stat_flat_kernel<<<1, 256, 0, stream>>>(efb, 3 * B_ * D_, st + 6);
  conv_pool_kernel<<<B_, 256, 0, stream>>>(knb, efb, st + 6, bn_g, bn_b, pooled);

  // 10. h = relu(pooled @ fc^T + b)  (512 x 400)
  gemm(pooled, fc_w, fc_b, hb, 512, 400, 6400, 6400, 6400, 400, 1, 1);

  // 11. out = h @ ent_emb^T + bias  (512 x 200000)  -- the 82 GFLOP GEMM
  gemm(hb, ent_emb, biasv, out, 512, N_ENT, 400, 400, 400, N_ENT, 1, 0);
}
